// MultiHeadSelfAttention_14989435863450
// MI455X (gfx1250) — matrix operand order repaired and checked by reading, not compiled, not run on hardware
//
#include <hip/hip_runtime.h>
#include <hip/hip_bf16.h>

#define DMODEL 512
#define NHEADS 8
#define DHEAD  64
#define BATCH  2
#define SEQ    2048
#define MROWS  (BATCH * SEQ)

#define AS1 __attribute__((address_space(1)))
#define AS3 __attribute__((address_space(3)))

typedef __bf16 bf16_t;
typedef __attribute__((ext_vector_type(16))) __bf16 v16bf;
typedef __attribute__((ext_vector_type(8)))  float  v8f;
typedef __attribute__((ext_vector_type(4)))  int    v4i_t;

#if defined(__has_builtin)
#if __has_builtin(__builtin_amdgcn_global_load_async_to_lds_b128)
#define HAVE_ASYNC_LDS 1
#endif
#if __has_builtin(__builtin_amdgcn_s_wait_asynccnt)
#define HAVE_WAIT_ASYNC 1
#endif
#endif

// ---------------------------------------------------------------------------
// CDNA5 async global->LDS copy (16B), with wide-load fallback
// Builtin signature (from clang diagnostic): (int4 AS1*, int4 AS3*, imm, imm)
// ---------------------------------------------------------------------------
__device__ __forceinline__ void async_copy_b128(const void* g, void* l) {
#ifdef HAVE_ASYNC_LDS
  __builtin_amdgcn_global_load_async_to_lds_b128((AS1 v4i_t*)g, (AS3 v4i_t*)l, 0, 0);
#else
  *(int4*)l = *(const int4*)g;  // global_load_b128 + ds_store_b128
#endif
}
__device__ __forceinline__ void wait_async_lds() {
#ifdef HAVE_ASYNC_LDS
#ifdef HAVE_WAIT_ASYNC
  __builtin_amdgcn_s_wait_asynccnt(0);
#else
  asm volatile("s_wait_asynccnt 0" ::: "memory");
#endif
#endif
}

// ---------------------------------------------------------------------------
// WMMA helpers (gfx1250 wave32, 16x16x32 bf16 -> f32)
// ---------------------------------------------------------------------------
__device__ __forceinline__ v8f wmma_bf16(v16bf a, v16bf b, v8f c) {
  return __builtin_amdgcn_wmma_f32_16x16x32_bf16(
      false, a, false, b, (short)0, c, false, false);
}

// A fragment: 16(M) x 32(K), row-major source, leading dim lda (elements).
__device__ __forceinline__ v16bf load_a_frag(const bf16_t* base, int lda) {
  const int lane = threadIdx.x & 31;
  const int hw = lane >> 4;
  const int m = lane & 15;
  v16bf a;
#pragma unroll
  for (int e = 0; e < 16; ++e) {
    const int k = hw * 16 + e;
    a[e] = base[(size_t)m * lda + k];
  }
  return a;
}

// B fragment: 32(K) x 16(N). Element B[k][n] at base[k*stride_k + n*stride_n].
__device__ __forceinline__ v16bf load_b_frag(const bf16_t* base, int stride_k, int stride_n) {
  const int lane = threadIdx.x & 31;
  const int hw = lane >> 4;
  const int n = lane & 15;
  v16bf b;
#pragma unroll
  for (int e = 0; e < 16; ++e) {
    const int k = hw * 16 + e;
    b[e] = base[(size_t)k * stride_k + (size_t)n * stride_n];
  }
  return b;
}
// C/D fragment element i: row M = hw*8 + i, col N = lane&15.

// ---------------------------------------------------------------------------
// fp32 -> bf16 cast
// ---------------------------------------------------------------------------
__global__ void cast_f32_to_bf16(const float* __restrict__ in, bf16_t* __restrict__ out, int n) {
  int i = blockIdx.x * blockDim.x + threadIdx.x;
  const int stride = gridDim.x * blockDim.x;
  for (; i < n; i += stride) out[i] = (bf16_t)in[i];
}

// ---------------------------------------------------------------------------
// Tiled WMMA GEMM: C[M,512] = A[M,512] @ W[512,512] (+ bias / rowvec epilogue)
// Block = 256 threads (8 waves) computing a 128x64 tile; LDS-staged A & W.
// MODE 0: bf16 out1, optional bias
// MODE 1: bias + (out1 = +add1, out2 = +add2)   (q+u / q+v)
// MODE 2: f32 outf + bias                        (final projection)
// ---------------------------------------------------------------------------
template <int MODE>
__global__ void gemm_wmma(const bf16_t* __restrict__ A, const bf16_t* __restrict__ W,
                          const float* __restrict__ bias,
                          const float* __restrict__ add1, const float* __restrict__ add2,
                          bf16_t* __restrict__ out1, bf16_t* __restrict__ out2,
                          float* __restrict__ outf) {
  __shared__ __align__(16) bf16_t Atile[128 * 32];
  __shared__ __align__(16) bf16_t Btile[32 * 64];
  const int m0 = blockIdx.x * 128;
  const int n0 = blockIdx.y * 64;
  const int tid = threadIdx.x;
  const int wave = tid >> 5;
  const int lane = tid & 31, hw = lane >> 4, nn = lane & 15;
  v8f acc[4] = {};

  const int ar = tid >> 2, aq = tid & 3;  // A: 128 rows x 4 16B segs (2 rows/thread)
  const int br = tid >> 3, bq = tid & 7;  // W: 32 rows x 8 16B segs

  for (int k0 = 0; k0 < DMODEL; k0 += 32) {
    __syncthreads();  // previous iteration's fragment reads must finish
    async_copy_b128(A + (size_t)(m0 + ar) * DMODEL + k0 + aq * 8, &Atile[ar * 32 + aq * 8]);
    async_copy_b128(A + (size_t)(m0 + 64 + ar) * DMODEL + k0 + aq * 8,
                    &Atile[(64 + ar) * 32 + aq * 8]);
    async_copy_b128(W + (size_t)(k0 + br) * DMODEL + n0 + bq * 8, &Btile[br * 64 + bq * 8]);
    wait_async_lds();
    __syncthreads();
    const v16bf a = load_a_frag(&Atile[(wave * 16) * 32], 32);
#pragma unroll
    for (int t = 0; t < 4; ++t)
      acc[t] = wmma_bf16(a, load_b_frag(&Btile[t * 16], 64, 1), acc[t]);
  }

  const int mrow0 = m0 + wave * 16;
#pragma unroll
  for (int t = 0; t < 4; ++t)
#pragma unroll
    for (int i = 0; i < 8; ++i) {
      const int row = mrow0 + hw * 8 + i;
      const int col = n0 + t * 16 + nn;
      float val = acc[t][i];
      if constexpr (MODE == 2) {
        outf[(size_t)row * DMODEL + col] = val + bias[col];
      } else if constexpr (MODE == 1) {
        val += bias[col];
        out1[(size_t)row * DMODEL + col] = (bf16_t)(val + add1[col]);
        out2[(size_t)row * DMODEL + col] = (bf16_t)(val + add2[col]);
      } else {
        if (bias) val += bias[col];
        out1[(size_t)row * DMODEL + col] = (bf16_t)val;
      }
    }
}

// ---------------------------------------------------------------------------
// pos_score[b,h,q,k] = dot(qv[b,q,h,:], p[b,k,h,:])   (bf16, pre-shift)
// ---------------------------------------------------------------------------
__global__ void pos_score_kernel(const bf16_t* __restrict__ qv, const bf16_t* __restrict__ pproj,
                                 bf16_t* __restrict__ ps) {
  const int q0 = blockIdx.x * 16;
  const int k0 = blockIdx.y * 16;
  const int bh = blockIdx.z;
  const int b = bh / NHEADS, h = bh % NHEADS;
  const bf16_t* qbase = qv + ((size_t)b * SEQ + q0) * DMODEL + h * DHEAD;
  const bf16_t* pbase = pproj + ((size_t)b * SEQ + k0) * DMODEL + h * DHEAD;
  v8f acc = {};
  acc = wmma_bf16(load_a_frag(qbase, DMODEL), load_b_frag(pbase, 1, DMODEL), acc);
  acc = wmma_bf16(load_a_frag(qbase + 32, DMODEL), load_b_frag(pbase + 32, 1, DMODEL), acc);
  const int lane = threadIdx.x & 31, hw = lane >> 4, nn = lane & 15;
  bf16_t* dst = ps + (size_t)bh * SEQ * SEQ;
#pragma unroll
  for (int i = 0; i < 8; ++i)
    dst[(size_t)(q0 + hw * 8 + i) * SEQ + k0 + nn] = (bf16_t)acc[i];
}

// ---------------------------------------------------------------------------
// Attention: per (b,h, 16-row q tile), streaming softmax over all keys.
// rel_shift remap:  k<=q : pos[q, S-1-q+k] ; k==q+1 : 0 ; k>q+1 : pos[q+1, k-q-2]
// ---------------------------------------------------------------------------
__global__ void attn_kernel(const bf16_t* __restrict__ qu, const bf16_t* __restrict__ kbuf,
                            const bf16_t* __restrict__ vbuf, const bf16_t* __restrict__ posm,
                            bf16_t* __restrict__ attn_out) {
  __shared__ __align__(16) bf16_t Plds[16 * 32];
  __shared__ __align__(16) bf16_t Vlds[32 * 64];
  const int q0 = blockIdx.x * 16;
  const int bh = blockIdx.y;
  const int b = bh / NHEADS, h = bh % NHEADS;
  const int lane = threadIdx.x & 31, hw = lane >> 4, nn = lane & 15;

  const bf16_t* qbase = qu + ((size_t)b * SEQ + q0) * DMODEL + h * DHEAD;
  const v16bf aq0 = load_a_frag(qbase, DMODEL);
  const v16bf aq1 = load_a_frag(qbase + 32, DMODEL);
  const bf16_t* ps = posm + (size_t)bh * SEQ * SEQ;

  v8f O[4] = {};
  float mrow[8], lrow[8];
#pragma unroll
  for (int i = 0; i < 8; ++i) { mrow[i] = -3.0e38f; lrow[i] = 0.f; }
  const float scl = 0.044194173824159216f;  // 1/sqrt(512)

  for (int k0 = 0; k0 < SEQ; k0 += 32) {
    // --- stage V tile early (one 64-elem row per lane, 8 x 16B copies) ---
    asm volatile("s_wait_dscnt 0" ::: "memory");  // prior PV ds-reads done
    {
      const bf16_t* vrow = vbuf + ((size_t)b * SEQ + k0 + lane) * DMODEL + h * DHEAD;
#pragma unroll
      for (int s = 0; s < 8; ++s) async_copy_b128(vrow + s * 8, &Vlds[lane * 64 + s * 8]);
    }

    // --- content scores (q+u) @ k^T ---
    v8f sc[2] = {};
#pragma unroll
    for (int sub = 0; sub < 2; ++sub) {
      const int kk0 = k0 + sub * 16;
      const bf16_t* kb0 = kbuf + ((size_t)b * SEQ + kk0) * DMODEL + h * DHEAD;
      sc[sub] = wmma_bf16(aq0, load_b_frag(kb0, 1, DMODEL), sc[sub]);
      sc[sub] = wmma_bf16(aq1, load_b_frag(kb0 + 32, 1, DMODEL), sc[sub]);
#pragma unroll
      for (int i = 0; i < 8; ++i) {
        const int q = q0 + hw * 8 + i;
        const int k = kk0 + nn;
        float pv = 0.f;
        if (k <= q)         pv = (float)ps[(size_t)q * SEQ + (SEQ - 1 - q + k)];
        else if (k > q + 1) pv = (float)ps[(size_t)(q + 1) * SEQ + (k - q - 2)];
        sc[sub][i] = (sc[sub][i] + pv) * scl;
      }
    }

    // --- online softmax (row reductions across each 16-lane half) ---
    float cmax[8], csum[8], scale8[8];
#pragma unroll
    for (int i = 0; i < 8; ++i) cmax[i] = fmaxf(sc[0][i], sc[1][i]);
#pragma unroll
    for (int off = 1; off < 16; off <<= 1)
#pragma unroll
      for (int i = 0; i < 8; ++i) cmax[i] = fmaxf(cmax[i], __shfl_xor(cmax[i], off, 32));
#pragma unroll
    for (int i = 0; i < 8; ++i) {
      const float mn = fmaxf(mrow[i], cmax[i]);
      scale8[i] = __expf(mrow[i] - mn);
      mrow[i] = mn;
      csum[i] = 0.f;
    }
#pragma unroll
    for (int sub = 0; sub < 2; ++sub)
#pragma unroll
      for (int i = 0; i < 8; ++i) {
        const float p = __expf(sc[sub][i] - mrow[i]);
        sc[sub][i] = p;
        csum[i] += p;
      }
#pragma unroll
    for (int off = 1; off < 16; off <<= 1)
#pragma unroll
      for (int i = 0; i < 8; ++i) csum[i] += __shfl_xor(csum[i], off, 32);
#pragma unroll
    for (int i = 0; i < 8; ++i) lrow[i] = lrow[i] * scale8[i] + csum[i];

    // --- P (C-layout) -> A fragment via LDS; per-wave DS is in-order ---
#pragma unroll
    for (int sub = 0; sub < 2; ++sub)
#pragma unroll
      for (int i = 0; i < 8; ++i)
        Plds[(hw * 8 + i) * 32 + sub * 16 + nn] = (bf16_t)sc[sub][i];
    const v16bf pa = load_a_frag(Plds, 32);

    // --- O = O*scale + P @ V ---
    wait_async_lds();
#pragma unroll
    for (int t = 0; t < 4; ++t) {
#pragma unroll
      for (int i = 0; i < 8; ++i) O[t][i] *= scale8[i];
      O[t] = wmma_bf16(pa, load_b_frag(&Vlds[t * 16], 64, 1), O[t]);
    }
  }

#pragma unroll
  for (int t = 0; t < 4; ++t)
#pragma unroll
    for (int i = 0; i < 8; ++i) {
      const int row = q0 + hw * 8 + i;
      const int col = h * DHEAD + t * 16 + nn;
      attn_out[((size_t)b * SEQ + row) * DMODEL + col] = (bf16_t)(O[t][i] / lrow[i]);
    }
}

// ---------------------------------------------------------------------------
// Host launcher
// ---------------------------------------------------------------------------
extern "C" void kernel_launch(void* const* d_in, const int* in_sizes, int n_in,
                              void* d_out, int out_size, void* d_ws, size_t ws_size,
                              hipStream_t stream) {
  (void)in_sizes; (void)n_in; (void)out_size; (void)ws_size;
  const float* x  = (const float*)d_in[0];
  const float* pe = (const float*)d_in[1];
  const float* Wq = (const float*)d_in[2];
  const float* bq = (const float*)d_in[3];
  const float* Wk = (const float*)d_in[4];
  const float* bk = (const float*)d_in[5];
  const float* Wv = (const float*)d_in[6];
  const float* bv = (const float*)d_in[7];
  const float* Wp = (const float*)d_in[8];
  const float* u  = (const float*)d_in[9];
  const float* v  = (const float*)d_in[10];
  const float* Wo = (const float*)d_in[11];
  const float* bo = (const float*)d_in[12];
  float* out = (float*)d_out;

  char* ws = (char*)d_ws;
  size_t off = 0;
  auto alloc = [&](size_t bytes) -> void* {
    void* p = ws + off;
    off += (bytes + 255) & ~(size_t)255;
    return p;
  };
  const size_t actN = (size_t)MROWS * DMODEL;
  const size_t wN   = (size_t)DMODEL * DMODEL;
  bf16_t* xb   = (bf16_t*)alloc(actN * 2);
  bf16_t* pb   = (bf16_t*)alloc(actN * 2);
  bf16_t* Wqb  = (bf16_t*)alloc(wN * 2);
  bf16_t* Wkb  = (bf16_t*)alloc(wN * 2);
  bf16_t* Wvb  = (bf16_t*)alloc(wN * 2);
  bf16_t* Wpb  = (bf16_t*)alloc(wN * 2);
  bf16_t* Wob  = (bf16_t*)alloc(wN * 2);
  bf16_t* qub  = (bf16_t*)alloc(actN * 2);
  bf16_t* qvb  = (bf16_t*)alloc(actN * 2);
  bf16_t* kb   = (bf16_t*)alloc(actN * 2);
  bf16_t* vb   = (bf16_t*)alloc(actN * 2);
  bf16_t* ppb  = (bf16_t*)alloc(actN * 2);
  bf16_t* attb = (bf16_t*)alloc(actN * 2);
  bf16_t* posb = (bf16_t*)alloc((size_t)BATCH * NHEADS * SEQ * SEQ * 2);  // 128 MB

  // 1) casts
  cast_f32_to_bf16<<<4096, 256, 0, stream>>>(x,  xb, (int)actN);
  cast_f32_to_bf16<<<4096, 256, 0, stream>>>(pe, pb, (int)actN);
  cast_f32_to_bf16<<<1024, 256, 0, stream>>>(Wq, Wqb, (int)wN);
  cast_f32_to_bf16<<<1024, 256, 0, stream>>>(Wk, Wkb, (int)wN);
  cast_f32_to_bf16<<<1024, 256, 0, stream>>>(Wv, Wvb, (int)wN);
  cast_f32_to_bf16<<<1024, 256, 0, stream>>>(Wp, Wpb, (int)wN);
  cast_f32_to_bf16<<<1024, 256, 0, stream>>>(Wo, Wob, (int)wN);

  // 2) projections: 8-wave blocks, 128x64 tiles, LDS-staged
  dim3 ggrid(MROWS / 128, DMODEL / 64);
  gemm_wmma<1><<<ggrid, 256, 0, stream>>>(xb, Wqb, bq, u, v, qub, qvb, nullptr);
  gemm_wmma<0><<<ggrid, 256, 0, stream>>>(xb, Wkb, bk, nullptr, nullptr, kb, nullptr, nullptr);
  gemm_wmma<0><<<ggrid, 256, 0, stream>>>(xb, Wvb, bv, nullptr, nullptr, vb, nullptr, nullptr);
  gemm_wmma<0><<<ggrid, 256, 0, stream>>>(pb, Wpb, nullptr, nullptr, nullptr, ppb, nullptr, nullptr);

  // 3) full positional score matrix (pre-shift)
  dim3 pgrid(SEQ / 16, SEQ / 16, BATCH * NHEADS);
  pos_score_kernel<<<pgrid, 32, 0, stream>>>(qvb, ppb, posb);

  // 4) attention with rel-shift gather + online softmax + PV
  dim3 agrid(SEQ / 16, BATCH * NHEADS);
  attn_kernel<<<agrid, 32, 0, stream>>>(qub, kb, vb, posb, attb);

  // 5) output projection -> fp32
  gemm_wmma<2><<<ggrid, 256, 0, stream>>>(attb, Wob, bo, nullptr, nullptr, nullptr, nullptr, out);
}
